// MyRNN_34462817583398
// MI455X (gfx1250) — matrix operand order repaired and checked by reading, not compiled, not run on hardware
//
#include <hip/hip_runtime.h>

// ---- types ------------------------------------------------------------
typedef __attribute__((ext_vector_type(16))) __bf16  v16bf;
typedef __attribute__((ext_vector_type(8)))  float   v8f;
typedef __attribute__((ext_vector_type(4)))  unsigned int v4u;

union Frag {          // 32 bytes: one WMMA bf16 operand fragment per lane
    v4u   q[2];
    v16bf v;
};

__device__ __forceinline__ unsigned int f2bf(float f) {
    union { float f; unsigned int u; } c; c.f = f;
    unsigned int u = c.u;
    return (u + 0x7FFFu + ((u >> 16) & 1u)) >> 16;   // RNE
}
__device__ __forceinline__ unsigned int pack2(float a, float b) {
    return f2bf(a) | (f2bf(b) << 16);
}
__device__ __forceinline__ float sigmoidf(float x) {
    return 1.0f / (1.0f + __expf(-x));
}

// Dimensions
#define B_  64
#define T_  128
#define E_  300
#define H_  512
#define P_  3
#define M_  (B_*T_)      // 8192

// ---- pack W [512x512] f32 -> bf16 B-fragments -------------------------
// B-frag layout (16x16x32 bf16): lane l, g=l>>4, n = nt*16+(l&15);
// dword j holds W[kt*32 + g*8 + 2j + (j>=4)*8][n] (lo) and [.. +1][n] (hi).
__global__ void pack_w_kernel(const float* __restrict__ W, unsigned int* __restrict__ Wpk) {
    int id = blockIdx.x * 256 + threadIdx.x;          // 16*32*32*8 = 131072
    int j    = id & 7;
    int lane = (id >> 3) & 31;
    int f    = id >> 8;                               // kt*32 + nt
    int kt = f >> 5, nt = f & 31;
    int g = lane >> 4;
    int n = nt * 16 + (lane & 15);
    int k = kt * 32 + ((j < 4) ? (g * 8 + 2 * j) : (16 + g * 8 + 2 * (j - 4)));
    Wpk[id] = pack2(W[k * H_ + n], W[(k + 1) * H_ + n]);
}

// ---- pack U [300x512] f32 -> bf16 B-fragments, K padded to 320 --------
__global__ void pack_u_kernel(const float* __restrict__ U, unsigned int* __restrict__ Upk) {
    int id = blockIdx.x * 256 + threadIdx.x;          // 10*32*32*8 = 81920
    int j    = id & 7;
    int lane = (id >> 3) & 31;
    int f    = id >> 8;                               // kt*32 + nt
    int kt = f >> 5, nt = f & 31;
    int g = lane >> 4;
    int n = nt * 16 + (lane & 15);
    int k = kt * 32 + ((j < 4) ? (g * 8 + 2 * j) : (16 + g * 8 + 2 * (j - 4)));
    float v0 = (k     < E_) ? U[k * H_ + n]       : 0.0f;
    float v1 = (k + 1 < E_) ? U[(k + 1) * H_ + n] : 0.0f;
    Upk[id] = pack2(v0, v1);
}

// ---- gather emb rows, pack to bf16 A-fragments ------------------------
// A-frag layout (16x32 bf16): lane l, g=l>>4, m = mt*16+(l&15);
// dwords 0..3 hold K = kt*32 + g*8 + 2j (+1); dwords 4..7 hold K = +16.
__global__ void pack_x_kernel(const int* __restrict__ text, const float* __restrict__ emb,
                              unsigned int* __restrict__ Apk) {
    int id = blockIdx.x * 256 + threadIdx.x;          // 512*10*32*8 = 1310720
    int j    = id & 7;
    int lane = (id >> 3) & 31;
    int f    = id >> 8;                               // mt*10 + kt
    int kt = f % 10, mt = f / 10;
    int g = lane >> 4;
    int m = mt * 16 + (lane & 15);
    int row = text[(m >> 7) * T_ + (m & (T_ - 1))];   // text[b][t], m = b*T + t
    int koff = (j < 4) ? (g * 8 + 2 * j) : (16 + g * 8 + 2 * (j - 4));
    int k = kt * 32 + koff;
    float v0 = (k     < E_) ? emb[row * E_ + k]     : 0.0f;
    float v1 = (k + 1 < E_) ? emb[row * E_ + k + 1] : 0.0f;
    Apk[id] = pack2(v0, v1);
}

// ---- xU GEMM: p[t][b][n] = (emb[text] @ U)[b*T+t][n] + bias[n] --------
// One wave per 16x16 output tile; M=8192 (512 mtiles), N=512 (32 ntiles), K=320.
__global__ __launch_bounds__(256) void xu_gemm_kernel(const unsigned int* __restrict__ Apk,
                                                      const unsigned int* __restrict__ Upk,
                                                      const float* __restrict__ bias,
                                                      float* __restrict__ p) {
    int gw   = blockIdx.x * 8 + (threadIdx.x >> 5);   // 0..16383
    int lane = threadIdx.x & 31;
    int mt = gw >> 5;                                 // 0..511
    int nt = gw & 31;                                 // 0..31

    // Base pointers; all per-kt strides are compile-time constants.
    const char* abase = (const char*)Apk + (size_t)(mt * 10 * 32 + lane) * 32; // +kt*1024
    const char* bbase = (const char*)Upk + (size_t)(nt * 32 + lane) * 32;      // +kt*32768

    v8f acc = {};
#pragma unroll
    for (int kt = 0; kt < 10; ++kt) {
        Frag a, b;
        a.q[0] = *(const v4u*)(abase + kt * 1024);
        a.q[1] = *(const v4u*)(abase + kt * 1024 + 16);
        b.q[0] = *(const v4u*)(bbase + kt * 32768);
        b.q[1] = *(const v4u*)(bbase + kt * 32768 + 16);
        acc = __builtin_amdgcn_wmma_f32_16x16x32_bf16(false, a.v, false, b.v,
                                                      (short)0, acc, false, false);
    }
    int g = lane >> 4, lm = lane & 15;
    int n = nt * 16 + lm;
    float bv = bias[n];
#pragma unroll
    for (int r = 0; r < 8; ++r) {
        int m  = mt * 16 + r + 8 * g;                 // m = b*T + t
        int bb = m >> 7, tt = m & (T_ - 1);
        p[(tt * B_ + bb) * H_ + n] = acc[r] + bv;     // store as [T][B][H]
    }
}

// ---- sequential scan: one WGP, 32 waves, h in LDS as bf16 -------------
// Step: h = sigmoid(p_t + h@W), t = 0..127; then y = sigmoid(p_127 + h@W).
__global__ __launch_bounds__(1024, 1) void rnn_scan_kernel(const float* __restrict__ p,
                                                           const unsigned int* __restrict__ Wpk,
                                                           float* __restrict__ y) {
    __shared__ __align__(16) unsigned short hbuf[B_ * H_];   // 64 KB bf16
    const int tid  = threadIdx.x;
    const int lane = tid & 31;
    const int w    = tid >> 5;          // 0..31
    const int mt   = w >> 3;            // 0..3  (rows 16*mt..)
    const int nt0  = (w & 7) * 4;       // 4 n-tiles per wave
    const int g    = lane >> 4;
    const int lm   = lane & 15;

    for (int i = tid; i < B_ * H_; i += 1024) hbuf[i] = 0;
    __syncthreads();

    // A fragments: base = row (mt*16+lm) of hbuf, + g*16 bytes; per-kt
    // stride 64 bytes, second half at +32 bytes (all constant offsets).
    const char* hbase = (const char*)&hbuf[(mt * 16 + lm) * H_] + g * 16;
    // B fragments: base for (kt=0, nt0); kt stride 32 KB, n-tile stride 1 KB.
    const char* wbase = (const char*)Wpk + (size_t)(nt0 * 32 + lane) * 32;
    // p / y element (j,r): base + r*512 floats + j*16 floats (constant).
    const float* prow0 = p + (size_t)(mt * 16 + 8 * g) * H_ + nt0 * 16 + lm;
    float*       yrow  = y + (size_t)(mt * 16 + 8 * g) * H_ + nt0 * 16 + lm;

    for (int t = 0; t < T_ + 1; ++t) {
        const int pt = (t < T_) ? t : (T_ - 1);
        const float* pb = prow0 + (size_t)pt * (B_ * H_);

        // Issue p_t loads first; latency hidden under the 64-WMMA GEMM.
        float pv[4][8];
#pragma unroll
        for (int j = 0; j < 4; ++j)
#pragma unroll
            for (int r = 0; r < 8; ++r)
                pv[j][r] = pb[r * H_ + j * 16];

        v8f acc[4] = {{}, {}, {}, {}};
#pragma unroll
        for (int kt = 0; kt < 16; ++kt) {
            Frag a;
            a.q[0] = *(const v4u*)(hbase + kt * 64);
            a.q[1] = *(const v4u*)(hbase + kt * 64 + 32);
#pragma unroll
            for (int j = 0; j < 4; ++j) {
                Frag b;
                const char* bp = wbase + kt * 32768 + j * 1024;
                b.q[0] = *(const v4u*)bp;
                b.q[1] = *(const v4u*)(bp + 16);
                acc[j] = __builtin_amdgcn_wmma_f32_16x16x32_bf16(false, a.v, false, b.v,
                                                                 (short)0, acc[j], false, false);
            }
        }
        __syncthreads();                 // all reads of hbuf complete
#pragma unroll
        for (int j = 0; j < 4; ++j) {
            const int n = (nt0 + j) * 16 + lm;
#pragma unroll
            for (int r = 0; r < 8; ++r) {
                const int bb = mt * 16 + r + 8 * g;
                float v = sigmoidf(acc[j][r] + pv[j][r]);
                if (t < T_) hbuf[bb * H_ + n] = (unsigned short)f2bf(v);
                else        yrow[r * H_ + j * 16] = v;
            }
        }
        __syncthreads();
    }
}

// ---- head: out[b][p] = y[b] . Wd[:,p] + bd[p] -------------------------
__global__ void head_kernel(const float* __restrict__ y, const float* __restrict__ Wd,
                            const float* __restrict__ bd, float* __restrict__ out) {
    int id = threadIdx.x;                // 0..191
    int bb = id / P_, pc = id % P_;
    float s = bd[pc];
    for (int k = 0; k < H_; ++k) s += y[bb * H_ + k] * Wd[k * P_ + pc];
    out[id] = s;
}

// ---- launch -----------------------------------------------------------
extern "C" void kernel_launch(void* const* d_in, const int* in_sizes, int n_in,
                              void* d_out, int out_size, void* d_ws, size_t ws_size,
                              hipStream_t stream) {
    const int*   text = (const int*)  d_in[0];
    const float* emb  = (const float*)d_in[1];
    const float* U    = (const float*)d_in[2];
    const float* W    = (const float*)d_in[3];
    const float* bias = (const float*)d_in[4];
    const float* Wd   = (const float*)d_in[5];
    const float* bd   = (const float*)d_in[6];
    float* out = (float*)d_out;

    char* ws = (char*)d_ws;
    size_t o = 0;
    unsigned int* Wpk = (unsigned int*)(ws + o); o += (size_t)16 * 32 * 32 * 8 * 4;   // 512 KB
    unsigned int* Upk = (unsigned int*)(ws + o); o += (size_t)10 * 32 * 32 * 8 * 4;   // 320 KB
    unsigned int* Apk = (unsigned int*)(ws + o); o += (size_t)512 * 10 * 32 * 8 * 4;  // 5.24 MB
    float*        p   = (float*)      (ws + o); o += (size_t)M_ * H_ * 4;             // 16.8 MB
    float*        y   = (float*)      (ws + o); o += (size_t)B_ * H_ * 4;             // 128 KB

    pack_w_kernel<<<512, 256, 0, stream>>>(W, Wpk);
    pack_u_kernel<<<320, 256, 0, stream>>>(U, Upk);
    pack_x_kernel<<<5120, 256, 0, stream>>>(text, emb, Apk);
    xu_gemm_kernel<<<2048, 256, 0, stream>>>(Apk, Upk, bias, p);
    rnn_scan_kernel<<<1, 1024, 0, stream>>>(p, Wpk, y);
    head_kernel<<<1, 192, 0, stream>>>(y, Wd, bd, out);
}
